// ChannelAttentionBlock_18227841204234
// MI455X (gfx1250) — matrix operand order repaired and checked by reading, not compiled, not run on hardware
//
#include <hip/hip_runtime.h>

// ---------------------------------------------------------------------------
// ChannelAttentionBlock for MI455X (gfx1250, wave32, WMMA bf16 f32-accum)
// x: [16, 512, 64, 64] f32.  S = Q Q^T, attn = softmax(max-S), O = attn Q,
// out = gamma*O + x.  GEMMs on v_wmma_f32_16x16x32_bf16 with double-buffered
// CDNA5 async global->LDS staging (global_load_async_to_lds_b128 / ASYNCcnt)
// overlapping the WMMA pipe.
// ---------------------------------------------------------------------------

typedef __attribute__((ext_vector_type(16))) __bf16 v16bf;
typedef __attribute__((ext_vector_type(8)))  float  v8f;

#define BATCH 16
#define CCH   512      // channels
#define NSP   4096     // H*W
#define TILE  128      // output tile per block (both GEMMs)
#define KCH   64       // K staged per LDS round
#define LDT   72       // padded LDS row stride in bf16 elems (16B-aligned)

__device__ __forceinline__ unsigned short f32_to_bf16(float f) {
  unsigned int u = __float_as_uint(f);
  unsigned int r = u + 0x7FFFu + ((u >> 16) & 1u);  // round-to-nearest-even
  return (unsigned short)(r >> 16);
}

// Flat aperture rule: for LDS, addr[31:0] IS the LDS byte address.
__device__ __forceinline__ unsigned lds_off(const void* p) {
  return (unsigned)(unsigned long long)p;
}

// CDNA5 async global->LDS copy, 16B per lane, tracked by ASYNCcnt.
__device__ __forceinline__ void async_load_b128(unsigned lds_byte_off,
                                                const void* gptr) {
  asm volatile("global_load_async_to_lds_b128 %0, %1, off"
               :: "v"(lds_byte_off), "v"((unsigned long long)gptr)
               : "memory");
}
__device__ __forceinline__ void wait_async0() {
  asm volatile("s_wait_asynccnt 0" ::: "memory");
}

// ---------------------------------------------------------------- convert ---
__global__ void cvt_bf16_kernel(const float* __restrict__ x,
                                unsigned short* __restrict__ qb) {
  size_t i = ((size_t)blockIdx.x * blockDim.x + threadIdx.x) * 4;
  float4 f = *(const float4*)(x + i);
  unsigned int lo = (unsigned int)f32_to_bf16(f.x) | ((unsigned int)f32_to_bf16(f.y) << 16);
  unsigned int hi = (unsigned int)f32_to_bf16(f.z) | ((unsigned int)f32_to_bf16(f.w) << 16);
  uint2 p; p.x = lo; p.y = hi;
  *(uint2*)(qb + i) = p;
}

// ------------------------------------------------------- GEMM1: S = Q Q^T ---
// grid: (16 tiles, 16 batches), block 256 (8 waves). Double-buffered LDS.
__global__ __launch_bounds__(256)
void gram_kernel(const unsigned short* __restrict__ qb,
                 float* __restrict__ aff) {
  const int b  = blockIdx.y;
  const int m0 = (blockIdx.x >> 2) * TILE;
  const int n0 = (blockIdx.x & 3)  * TILE;

  __shared__ unsigned short sA[2][TILE * LDT];
  __shared__ unsigned short sB[2][TILE * LDT];

  const unsigned short* Q = qb + (size_t)b * CCH * NSP;

  const int tid  = threadIdx.x;
  const int lane = tid & 31;
  const int wave = tid >> 5;
  const int wr   = wave & 3;    // 4 row groups of 32
  const int wc   = wave >> 2;   // 2 col groups of 64

  v8f acc[2][4] = {};

  const int Mlane = lane & 15;
  const int kbA   = (lane < 16) ? 0 : 8;    // A-frag K base (16x32 bf16 layout)
  const int kbB   = (lane < 16) ? 0 : 8;    // B-frag K base (same layout as A)

  // Stage issue: 4 chunks/thread/tile; coordinates fixed per thread.
  auto issue = [&](int k0, int buf) {
#pragma unroll
    for (int i = 0; i < 4; ++i) {
      int idx = tid + i * 256;          // 1024 x 16B chunks per tile
      int row = idx >> 3;
      int c8  = (idx & 7) * 8;
      async_load_b128(lds_off(&sA[buf][row * LDT + c8]),
                      &Q[(size_t)(m0 + row) * NSP + k0 + c8]);
      async_load_b128(lds_off(&sB[buf][row * LDT + c8]),
                      &Q[(size_t)(n0 + row) * NSP + k0 + c8]);
    }
  };

  const int NSTAGE = NSP / KCH;   // 64
  issue(0, 0);
  wait_async0();
  __syncthreads();

  int cur = 0;
  for (int s = 0; s < NSTAGE; ++s) {
    if (s + 1 < NSTAGE) issue((s + 1) * KCH, cur ^ 1);   // overlap with WMMA

#pragma unroll
    for (int kk = 0; kk < KCH; kk += 32) {
      v16bf afr[2], bfr[4];
#pragma unroll
      for (int mt = 0; mt < 2; ++mt) {
        int row = wr * 32 + mt * 16 + Mlane;
        uint4* ap = (uint4*)&afr[mt];
        ap[0] = *(const uint4*)&sA[cur][row * LDT + kk + kbA];        // K kb..kb+7
        ap[1] = *(const uint4*)&sA[cur][row * LDT + kk + kbA + 16];   // K kb+16..kb+23
      }
#pragma unroll
      for (int nt = 0; nt < 4; ++nt) {
        int row = wc * 64 + nt * 16 + Mlane;   // B element (K,N) = Q[n0+N][k+K]
        uint4* bp = (uint4*)&bfr[nt];
        bp[0] = *(const uint4*)&sB[cur][row * LDT + kk + kbB];        // K kb..kb+7
        bp[1] = *(const uint4*)&sB[cur][row * LDT + kk + kbB + 16];   // K kb+16..kb+23
      }
#pragma unroll
      for (int mt = 0; mt < 2; ++mt)
#pragma unroll
        for (int nt = 0; nt < 4; ++nt)
          acc[mt][nt] = __builtin_amdgcn_wmma_f32_16x16x32_bf16(
              false, afr[mt], false, bfr[nt], (short)0, acc[mt][nt], false, false);
    }

    if (s + 1 < NSTAGE) wait_async0();   // next-stage copies ran under compute
    __syncthreads();
    cur ^= 1;
  }

  // C/D layout: VGPR r -> M=r (lanes 0-15) or M=r+8 (lanes 16-31); N = lane&15
  const int nlane = lane & 15;
  const int moff  = (lane < 16) ? 0 : 8;
  float* S = aff + (size_t)b * CCH * CCH;
#pragma unroll
  for (int mt = 0; mt < 2; ++mt)
#pragma unroll
    for (int nt = 0; nt < 4; ++nt)
#pragma unroll
      for (int r = 0; r < 8; ++r) {
        int row = m0 + wr * 32 + mt * 16 + moff + r;
        int col = n0 + wc * 64 + nt * 16 + nlane;
        S[(size_t)row * CCH + col] = acc[mt][nt][r];
      }
}

// ------------------------------------------------ softmax(max - S) rows ----
// attn_ij = exp(min_i - S_ij) / sum_j exp(min_i - S_ij);  one block per row.
__global__ void softmax_kernel(const float* __restrict__ aff,
                               unsigned short* __restrict__ attn) {
  const float* a = aff + (size_t)blockIdx.x * CCH;
  unsigned short* o = attn + (size_t)blockIdx.x * CCH;
  __shared__ float red[256];
  const int t = threadIdx.x;

  float v0 = a[t], v1 = a[t + 256];
  red[t] = fminf(v0, v1);
  __syncthreads();
  for (int s = 128; s > 0; s >>= 1) {
    if (t < s) red[t] = fminf(red[t], red[t + s]);
    __syncthreads();
  }
  float mn = red[0];
  __syncthreads();

  float e0 = __expf(mn - v0), e1 = __expf(mn - v1);
  red[t] = e0 + e1;
  __syncthreads();
  for (int s = 128; s > 0; s >>= 1) {
    if (t < s) red[t] += red[t + s];
    __syncthreads();
  }
  float inv = 1.0f / red[0];
  o[t]       = f32_to_bf16(e0 * inv);
  o[t + 256] = f32_to_bf16(e1 * inv);
}

// ---------------------------------- GEMM2: out = gamma*(attn Q) + x --------
// grid: (128 tiles, 16 batches), block 256. Double-buffered LDS pipeline:
// async A-copies + global B-row loads issued before compute; B pack/store
// after compute, all covered by the stage barrier.
__global__ __launch_bounds__(256)
void attnmm_kernel(const unsigned short* __restrict__ attn,
                   const unsigned short* __restrict__ qb,
                   const float* __restrict__ x,
                   const float* __restrict__ gamma,
                   float* __restrict__ out) {
  const int b  = blockIdx.y;
  const int m0 = (blockIdx.x >> 5) * TILE;   // 512/128 = 4 row tiles
  const int n1 = (blockIdx.x & 31) * TILE;   // 4096/128 = 32 col tiles

  __shared__ unsigned short sA[2][TILE * LDT];    // attn tile, row-major [m][k]
  __shared__ unsigned short sBt[2][TILE * LDT];   // Q tile transposed: [n][k]

  const unsigned short* P = attn + (size_t)b * CCH * CCH;
  const unsigned short* Q = qb + (size_t)b * CCH * NSP;

  const int tid  = threadIdx.x;
  const int lane = tid & 31;
  const int wave = tid >> 5;
  const int wr   = wave & 3;
  const int wc   = wave >> 2;

  v8f acc[2][4] = {};

  const int Mlane = lane & 15;
  const int kbA   = (lane < 16) ? 0 : 8;
  const int kbB   = (lane < 16) ? 0 : 8;

  // Transpose micro-block coords: 4(k) x 8(n) per thread.
  const int tkb = (tid >> 4) * 4;       // 0..60, step 4
  const int tnb = (tid & 15) * 8;       // 0..120, step 8

  auto issue_a = [&](int k0, int buf) {
#pragma unroll
    for (int i = 0; i < 4; ++i) {
      int idx = tid + i * 256;
      int row = idx >> 3;
      int c8  = (idx & 7) * 8;
      async_load_b128(lds_off(&sA[buf][row * LDT + c8]),
                      &P[(size_t)(m0 + row) * CCH + k0 + c8]);
    }
  };
  auto load_b = [&](int k0, uint4* r) {
#pragma unroll
    for (int j = 0; j < 4; ++j)
      r[j] = *(const uint4*)&Q[(size_t)(k0 + tkb + j) * NSP + n1 + tnb];
  };
  auto store_bt = [&](const uint4* r, int buf) {
    const unsigned short* p0 = (const unsigned short*)&r[0];
    const unsigned short* p1 = (const unsigned short*)&r[1];
    const unsigned short* p2 = (const unsigned short*)&r[2];
    const unsigned short* p3 = (const unsigned short*)&r[3];
#pragma unroll
    for (int j = 0; j < 8; ++j) {
      uint2 v;
      v.x = (unsigned)p0[j] | ((unsigned)p1[j] << 16);
      v.y = (unsigned)p2[j] | ((unsigned)p3[j] << 16);
      *(uint2*)&sBt[buf][(tnb + j) * LDT + tkb] = v;   // 8B-aligned
    }
  };

  const int NSTAGE = CCH / KCH;   // 8

  // Prologue: stage 0.
  {
    uint4 r[4];
    issue_a(0, 0);
    load_b(0, r);
    store_bt(r, 0);
    wait_async0();
    __syncthreads();
  }

  int cur = 0;
  for (int s = 0; s < NSTAGE; ++s) {
    uint4 rn[4];
    if (s + 1 < NSTAGE) {
      issue_a((s + 1) * KCH, cur ^ 1);   // async, overlaps WMMA
      load_b((s + 1) * KCH, rn);         // vmem latency hidden under WMMA
    }

#pragma unroll
    for (int kk = 0; kk < KCH; kk += 32) {
      v16bf afr[2], bfr[4];
#pragma unroll
      for (int mt = 0; mt < 2; ++mt) {
        int row = wr * 32 + mt * 16 + Mlane;
        uint4* ap = (uint4*)&afr[mt];
        ap[0] = *(const uint4*)&sA[cur][row * LDT + kk + kbA];
        ap[1] = *(const uint4*)&sA[cur][row * LDT + kk + kbA + 16];
      }
#pragma unroll
      for (int nt = 0; nt < 4; ++nt) {
        int row = wc * 64 + nt * 16 + Mlane;    // transposed: [n][k], K contiguous
        uint4* bp = (uint4*)&bfr[nt];
        bp[0] = *(const uint4*)&sBt[cur][row * LDT + kk + kbB];
        bp[1] = *(const uint4*)&sBt[cur][row * LDT + kk + kbB + 16];
      }
#pragma unroll
      for (int mt = 0; mt < 2; ++mt)
#pragma unroll
        for (int nt = 0; nt < 4; ++nt)
          acc[mt][nt] = __builtin_amdgcn_wmma_f32_16x16x32_bf16(
              false, afr[mt], false, bfr[nt], (short)0, acc[mt][nt], false, false);
    }

    if (s + 1 < NSTAGE) {
      store_bt(rn, cur ^ 1);   // sBt^1 free: last read in previous stage, before its barrier
      wait_async0();
    }
    __syncthreads();
    cur ^= 1;
  }

  const float g     = gamma[0];
  const int   nlane = lane & 15;
  const int   moff  = (lane < 16) ? 0 : 8;
#pragma unroll
  for (int mt = 0; mt < 2; ++mt)
#pragma unroll
    for (int nt = 0; nt < 4; ++nt)
#pragma unroll
      for (int r = 0; r < 8; ++r) {
        int row = m0 + wr * 32 + mt * 16 + moff + r;
        int col = n1 + wc * 64 + nt * 16 + nlane;
        size_t idx = ((size_t)b * CCH + row) * (size_t)NSP + col;
        out[idx] = g * acc[mt][nt][r] + x[idx];
      }
}

// ---------------------------------------------------------------------------
extern "C" void kernel_launch(void* const* d_in, const int* in_sizes, int n_in,
                              void* d_out, int out_size, void* d_ws, size_t ws_size,
                              hipStream_t stream) {
  const float* x     = (const float*)d_in[0];
  const float* gamma = (const float*)d_in[1];
  float*       out   = (float*)d_out;

  const size_t nelem    = (size_t)BATCH * CCH * NSP;      // 33,554,432
  const size_t qb_bytes = nelem * sizeof(unsigned short); // 64 MiB
  const size_t aff_bytes = (size_t)BATCH * CCH * CCH * sizeof(float); // 16 MiB

  unsigned short* qb   = (unsigned short*)d_ws;
  float*          aff  = (float*)((char*)d_ws + qb_bytes);
  unsigned short* attn = (unsigned short*)((char*)d_ws + qb_bytes + aff_bytes);

  // 1) f32 -> bf16
  cvt_bf16_kernel<<<(int)(nelem / 4 / 256), 256, 0, stream>>>(x, qb);
  // 2) Gram matrix S = Q Q^T  (bf16 WMMA, async double-buffered staging)
  gram_kernel<<<dim3(16, BATCH), 256, 0, stream>>>(qb, aff);
  // 3) row softmax of (max - S), emit bf16 attn
  softmax_kernel<<<BATCH * CCH, 256, 0, stream>>>(aff, attn);
  // 4) out = gamma * (attn @ Q) + x  (bf16 WMMA, fused epilogue)
  attnmm_kernel<<<dim3(128, BATCH), 256, 0, stream>>>(attn, qb, x, gamma, out);
}
